// MobilityPatternExtractor_34806414967142
// MI455X (gfx1250) — matrix operand order repaired and checked by reading, not compiled, not run on hardware
//
#include <hip/hip_runtime.h>
#include <hip/hip_bf16.h>

typedef __attribute__((ext_vector_type(16))) _Float16 v16h;
typedef __attribute__((ext_vector_type(8)))  float    v8f;

#define DAY_F 86400.0f

// ---------------- wave32 reduction helpers ----------------
__device__ __forceinline__ float wsumf(float v) {
#pragma unroll
  for (int o = 16; o > 0; o >>= 1) v += __shfl_xor(v, o, 32);
  return v;
}
__device__ __forceinline__ float wmaxf_red(float v) {
#pragma unroll
  for (int o = 16; o > 0; o >>= 1) v = fmaxf(v, __shfl_xor(v, o, 32));
  return v;
}
__device__ __forceinline__ float wminf_red(float v) {
#pragma unroll
  for (int o = 16; o > 0; o >>= 1) v = fminf(v, __shfl_xor(v, o, 32));
  return v;
}

__device__ __forceinline__ int lower_bound(const int* __restrict__ a, int n, int key) {
  int lo = 0, hi = n;
  while (lo < hi) {
    int mid = (lo + hi) >> 1;
    if (a[mid] < key) lo = mid + 1; else hi = mid;
  }
  return lo;
}

// Small branch MLP: din -> 16 (relu) -> 16. Lane-parallel over 16 outputs,
// hidden activations exchanged with width-16 shuffles.
template <int DIN>
__device__ __forceinline__ float branch_mlp(const float* x,
                                            const float* __restrict__ w1,
                                            const float* __restrict__ b1,
                                            const float* __restrict__ w2,
                                            const float* __restrict__ b2,
                                            int j) {
  float h = b1[j];
#pragma unroll
  for (int i = 0; i < DIN; ++i) h += x[i] * w1[i * 16 + j];
  h = fmaxf(h, 0.0f);
  float o = b2[j];
#pragma unroll
  for (int k = 0; k < 16; ++k) o += __shfl(h, k, 16) * w2[k * 16 + j];
  return o;
}

// ---------------- Kernel A: per-user segment stats + branch MLPs ----------------
// One wave per user; segment_ids sorted -> contiguous range via binary search.
// All reductions register+shuffle, zero atomics (bandwidth-bound phase).
__global__ void __launch_bounds__(256)
user_stats_kernel(const float* __restrict__ loc,   // [T,2]
                  const float* __restrict__ ts,    // [T]
                  const int*   __restrict__ cat,   // [T]
                  const int*   __restrict__ seg,   // [T] sorted
                  int T, int N,
                  const float* __restrict__ wl1, const float* __restrict__ bl1,
                  const float* __restrict__ wl2, const float* __restrict__ bl2,
                  const float* __restrict__ wt1, const float* __restrict__ bt1,
                  const float* __restrict__ wt2, const float* __restrict__ bt2,
                  const float* __restrict__ wa1, const float* __restrict__ ba1,
                  const float* __restrict__ wa2, const float* __restrict__ ba2,
                  const float* __restrict__ wm1, const float* __restrict__ bm1,
                  const float* __restrict__ wm2, const float* __restrict__ bm2,
                  float* __restrict__ feats)       // [N,64]
{
  const int wave = threadIdx.x >> 5;
  const int lane = threadIdx.x & 31;
  const int u = blockIdx.x * 8 + wave;
  if (u >= N) return;

  const int s = lower_bound(seg, T, u);
  const int e = lower_bound(seg, T, u + 1);
  const float L = (float)(e - s);          // >= 3 by construction
  const float invL = 1.0f / L;

  // ---- pass 1: means, histogram, min/max timestamp ----
  float sx = 0.f, sy = 0.f, sh = 0.f, sd = 0.f, sm = 0.f;
  float tmx = -3.402823466e38f, tmn = 3.402823466e38f;
  float hist[10];
#pragma unroll
  for (int k = 0; k < 10; ++k) hist[k] = 0.f;

  for (int i = s + lane; i < e; i += 32) {
    float x = loc[2 * i], y = loc[2 * i + 1];
    sx += x; sy += y;
    float t = ts[i];
    sh += fmodf(t, DAY_F) * (1.0f / 3600.0f);
    sd += fmodf(t * (1.0f / DAY_F), 7.0f);
    sm += fmodf(t * (1.0f / (30.0f * DAY_F)), 12.0f);
    tmx = fmaxf(tmx, t); tmn = fminf(tmn, t);
    int c = cat[i];
#pragma unroll
    for (int k = 0; k < 10; ++k) hist[k] += (c == k) ? 1.0f : 0.0f;
  }
  sx = wsumf(sx); sy = wsumf(sy);
  sh = wsumf(sh); sd = wsumf(sd); sm = wsumf(sm);
  tmx = wmaxf_red(tmx); tmn = wminf_red(tmn);
#pragma unroll
  for (int k = 0; k < 10; ++k) hist[k] = wsumf(hist[k]);

  const float hx = sx * invL, hy = sy * invL;
  const float tmp0 = sh * invL, tmp1 = sd * invL, tmp2 = sm * invL;

  // ---- pass 2: radius, inter-event diffs, unique locations ----
  float sdist = 0.f, sdist2 = 0.f, sdf = 0.f, sdf2 = 0.f, uq = 0.f;
  for (int i = s + lane; i < e; i += 32) {
    float x = loc[2 * i], y = loc[2 * i + 1];
    float dx = x - hx, dy = y - hy;
    float d = sqrtf(dx * dx + dy * dy);
    sdist += d; sdist2 += d * d;
    if (i + 1 < e) {
      float df = ts[i + 1] - ts[i];
      sdf += df; sdf2 += df * df;
    }
    bool isnew = true;
    for (int j = s; j < i; ++j) {
      if (loc[2 * j] == x && loc[2 * j + 1] == y) { isnew = false; break; }
    }
    uq += isnew ? 1.0f : 0.0f;
  }
  sdist = wsumf(sdist); sdist2 = wsumf(sdist2);
  sdf = wsumf(sdf);     sdf2 = wsumf(sdf2);
  uq = wsumf(uq);

  const float dmean = sdist * invL;
  const float radius = sqrtf(fmaxf((sdist2 - L * dmean * dmean) / (L - 1.0f), 0.0f));
  const float diversity = uq * invL;
  const float span = (tmx - tmn) * (1.0f / DAY_F);
  const float freq = L / fmaxf(span, 1.0f);
  const float m = L - 1.0f;
  const float dmu = sdf / m;
  const float dvar = (sdf2 - m * dmu * dmu) / (m - 1.0f);
  const float reg = 1.0f / (sqrtf(fmaxf(dvar, 0.0f)) + 1e-6f);

  // ---- branch MLPs ----
  const int j = lane & 15;
  float xin[10];
  float* fo = feats + (size_t)u * 64;

  xin[0] = hx; xin[1] = hy;
  float o = branch_mlp<2>(xin, wl1, bl1, wl2, bl2, j);
  if (lane < 16) fo[0 + j] = o;

  xin[0] = tmp0; xin[1] = tmp1; xin[2] = tmp2;
  o = branch_mlp<3>(xin, wt1, bt1, wt2, bt2, j);
  if (lane < 16) fo[16 + j] = o;

#pragma unroll
  for (int k = 0; k < 10; ++k) xin[k] = hist[k] * invL;   // hs == L > 0
  o = branch_mlp<10>(xin, wa1, ba1, wa2, ba2, j);
  if (lane < 16) fo[32 + j] = o;

  xin[0] = radius; xin[1] = diversity; xin[2] = freq; xin[3] = reg;
  o = branch_mlp<4>(xin, wm1, bm1, wm2, bm2, j);
  if (lane < 16) fo[48 + j] = o;
}

// ---------------- Kernel B: combined MLP (64->64 relu ->64) via WMMA ----------------
// Tile = 16 users x 64 feats per wave; 8x v_wmma_f32_16x16x32_f16 per layer.
// All operands are CONTIGUOUS 32B runs:
//  * weights pre-staged in B-operand layout -> v16h = 2x ds_load_b128
//  * layer-1 A loaded UNCONDITIONALLY (clamped row) as 4x global_load_b128/kt
//  * layer-1 output written directly in A-operand layout (permute on store side)
__global__ void __launch_bounds__(256)
cmb_mlp_wmma_kernel(const float* __restrict__ feats,  // [N,64]
                    const float* __restrict__ w1,     // [64,64]
                    const float* __restrict__ b1,     // [64]
                    const float* __restrict__ w2,     // [64,64]
                    const float* __restrict__ b2,     // [64]
                    float* __restrict__ out,          // [N,64]
                    int N)
{
  // B-operand staged weights: [kt][nt][lane][e] (8 KB each)
  __shared__ __align__(32) _Float16 sB1[2][4][32][16];
  __shared__ __align__(32) _Float16 sB2[2][4][32][16];
  // layer-1 activations already in A-operand layout: [wave][lane][kt*16+e]
  __shared__ __align__(32) _Float16 sHidA[8][32][32];
  __shared__ float sb1[64], sb2[64];

  const int tid = threadIdx.x;

  // Stage weights into operand layout: lane `li` of tile (kt,nt) needs
// B[K][col] with K = 32*kt + 8*(li>>4) + (e&7) + 16*(e>>3), col = 16*nt + (li&15).
  for (int i = tid; i < 4096; i += 256) {
    int e  = i & 15;
    int li = (i >> 4) & 31;
    int nt = (i >> 9) & 3;
    int kt = i >> 11;
    int K   = 32 * kt + 8 * (li >> 4) + (e & 7) + 16 * (e >> 3);
    int col = 16 * nt + (li & 15);
    sB1[kt][nt][li][e] = (_Float16)w1[K * 64 + col];
    sB2[kt][nt][li][e] = (_Float16)w2[K * 64 + col];
  }
  if (tid < 64) { sb1[tid] = b1[tid]; sb2[tid] = b2[tid]; }
  __syncthreads();

  const int wave = tid >> 5;
  const int lane = tid & 31;
  const int row  = lane & 15;
  const int half = lane >> 4;
  const int userBase = (blockIdx.x * 8 + wave) * 16;
  const int userA = userBase + row;
  // Clamp to a valid row: rows >= N read row N-1 (in-bounds); the resulting
  // garbage only reaches output rows >= N, whose stores are guarded below.
  const int userClamped = (userA < N) ? userA : (N - 1);

  v8f acc[4] = {};

  // ---- layer 1: unconditional vector loads ----
  const float* fr = feats + (size_t)userClamped * 64;
#pragma unroll
  for (int kt = 0; kt < 2; ++kt) {
    // lane needs K in two runs of 8 consecutive floats:
    //   [32*kt + 8*half .. +7]      -> a[0..7]
    //   [32*kt + 16 + 8*half .. +7] -> a[8..15]
    float4 q0 = *(const float4*)(fr + 32 * kt + 8 * half);
    float4 q1 = *(const float4*)(fr + 32 * kt + 8 * half + 4);
    float4 q2 = *(const float4*)(fr + 32 * kt + 16 + 8 * half);
    float4 q3 = *(const float4*)(fr + 32 * kt + 16 + 8 * half + 4);
    v16h a;
    a[0]  = (_Float16)q0.x; a[1]  = (_Float16)q0.y; a[2]  = (_Float16)q0.z; a[3]  = (_Float16)q0.w;
    a[4]  = (_Float16)q1.x; a[5]  = (_Float16)q1.y; a[6]  = (_Float16)q1.z; a[7]  = (_Float16)q1.w;
    a[8]  = (_Float16)q2.x; a[9]  = (_Float16)q2.y; a[10] = (_Float16)q2.z; a[11] = (_Float16)q2.w;
    a[12] = (_Float16)q3.x; a[13] = (_Float16)q3.y; a[14] = (_Float16)q3.z; a[15] = (_Float16)q3.w;
#pragma unroll
    for (int nt = 0; nt < 4; ++nt) {
      v16h bm = *(const v16h*)&sB1[kt][nt][lane][0];   // 2x ds_load_b128
      acc[nt] = __builtin_amdgcn_wmma_f32_16x16x32_f16(
          false, a, false, bm, (short)0, acc[nt], false, false);
    }
  }

  // bias + relu; write hidden DIRECTLY in A-operand layout (scatter on store).
  // For hidden element (M=row, K=col): kt=K>>5, k5=K&31, h=(k5>>3)&1,
  //   destLane = M + 16*h, e = 2*(4*((k5>>4)&1) + ((k5&7)>>1)) + (k5&1)
#pragma unroll
  for (int nt = 0; nt < 4; ++nt) {
    const int col = 16 * nt + row;
    const int kt2 = col >> 5;
    const int k5  = col & 31;
    const int h2  = (k5 >> 3) & 1;
    const int v2  = 4 * ((k5 >> 4) & 1) + ((k5 & 7) >> 1);
    const int e2  = 2 * v2 + (k5 & 1);
    const int slot = kt2 * 16 + e2;
#pragma unroll
    for (int g = 0; g < 8; ++g) {
      int M = g + 8 * half;
      float hval = fmaxf(acc[nt][g] + sb1[col], 0.0f);
      sHidA[wave][M + 16 * h2][slot] = (_Float16)hval;
    }
  }
  __syncthreads();

  // ---- layer 2: A operand is now a contiguous 32B run per lane ----
  v8f acc2[4] = {};
#pragma unroll
  for (int kt = 0; kt < 2; ++kt) {
    v16h a = *(const v16h*)&sHidA[wave][lane][kt * 16];  // 2x ds_load_b128
#pragma unroll
    for (int nt = 0; nt < 4; ++nt) {
      v16h bm = *(const v16h*)&sB2[kt][nt][lane][0];
      acc2[nt] = __builtin_amdgcn_wmma_f32_16x16x32_f16(
          false, a, false, bm, (short)0, acc2[nt], false, false);
    }
  }

  // bias + store: for fixed (nt,g) lanes 0..15 cover 16 consecutive cols of
  // one output row (coalesced 64B segments)
#pragma unroll
  for (int nt = 0; nt < 4; ++nt) {
#pragma unroll
    for (int g = 0; g < 8; ++g) {
      int M = g + 8 * half;
      int userO = userBase + M;
      if (userO < N) {
        int col = 16 * nt + row;
        out[(size_t)userO * 64 + col] = acc2[nt][g] + sb2[col];
      }
    }
  }
}

// ---------------- host launcher ----------------
extern "C" void kernel_launch(void* const* d_in, const int* in_sizes, int n_in,
                              void* d_out, int out_size, void* d_ws, size_t ws_size,
                              hipStream_t stream) {
  const float* locations  = (const float*)d_in[0];
  const float* timestamps = (const float*)d_in[1];
  const int*   categories = (const int*)d_in[2];
  const int*   seg        = (const int*)d_in[3];
  // d_in[4] = num_users (device scalar); N derived from out_size instead.
  const float* wl1 = (const float*)d_in[5],  *bl1 = (const float*)d_in[6];
  const float* wl2 = (const float*)d_in[7],  *bl2 = (const float*)d_in[8];
  const float* wt1 = (const float*)d_in[9],  *bt1 = (const float*)d_in[10];
  const float* wt2 = (const float*)d_in[11], *bt2 = (const float*)d_in[12];
  const float* wa1 = (const float*)d_in[13], *ba1 = (const float*)d_in[14];
  const float* wa2 = (const float*)d_in[15], *ba2 = (const float*)d_in[16];
  const float* wm1 = (const float*)d_in[17], *bm1 = (const float*)d_in[18];
  const float* wm2 = (const float*)d_in[19], *bm2 = (const float*)d_in[20];
  const float* wc1 = (const float*)d_in[21], *bc1 = (const float*)d_in[22];
  const float* wc2 = (const float*)d_in[23], *bc2 = (const float*)d_in[24];

  const int T = in_sizes[1];
  const int N = out_size / 64;
  float* feats = (float*)d_ws;             // N*64 floats of scratch
  float* out   = (float*)d_out;

  dim3 blk(256);
  dim3 grdA((N + 7) / 8);                   // 8 waves/block, 1 user/wave
  user_stats_kernel<<<grdA, blk, 0, stream>>>(
      locations, timestamps, categories, seg, T, N,
      wl1, bl1, wl2, bl2, wt1, bt1, wt2, bt2,
      wa1, ba1, wa2, ba2, wm1, bm1, wm2, bm2, feats);

  dim3 grdB((N + 127) / 128);               // 8 waves/block, 16 users/wave
  cmb_mlp_wmma_kernel<<<grdB, blk, 0, stream>>>(
      feats, wc1, bc1, wc2, bc2, out, N);
}
